// Model_VQVAE_10763188044264
// MI455X (gfx1250) — matrix operand order repaired and checked by reading, not compiled, not run on hardware
//
#include <hip/hip_runtime.h>
#include <hip/hip_bf16.h>

// ---------------------------------------------------------------------------
// CDNA5 (gfx1250) VQ-VAE forward.
// Core: implicit-GEMM conv kernel, wave32, 8 waves/block.
// Block tile 128x64, K-step 64; each wave owns a 32x32 C tile ->
// 8x v_wmma_f32_16x16x32_f16 per stage, double-buffered LDS, async
// global->LDS weight staging (ASYNCcnt) when available.
// ---------------------------------------------------------------------------

typedef __attribute__((ext_vector_type(16))) _Float16 v16h;
typedef __attribute__((ext_vector_type(8)))  _Float16 v8h;
typedef __attribute__((ext_vector_type(8)))  float    v8f;
typedef __attribute__((ext_vector_type(4)))  int      v4i;

#define TB 256
#define BM 128
#define BN 64
#define BK 64
#define LDA (BK + 8)   // 144B row stride: keeps 16B chunks aligned, skews banks
#define LDB (BK + 8)

static_assert(BM == 128 && BN == 64 && BK == 64 && TB == 256, "tile index math");

#if __has_builtin(__builtin_amdgcn_global_load_async_to_lds_b128)
#define USE_ASYNC_LDS 1
typedef __attribute__((address_space(1))) v4i* gptr_v4i;
typedef __attribute__((address_space(3))) v4i* lptr_v4i;
#endif

__device__ __forceinline__ void wait_async0()
{
#if __has_builtin(__builtin_amdgcn_s_wait_asynccnt)
    __builtin_amdgcn_s_wait_asynccnt(0);
#elif defined(USE_ASYNC_LDS)
    asm volatile("s_wait_asynccnt 0x0" ::: "memory");
#endif
}

// ---------------------------------------------------------------------------
// Implicit-GEMM convolution (also used for the VQ distance GEMM).
//   Y[M=(Nb*OH*OW), N=Cout] = im2col(X)[M, Kdim] * Wt^T   (Wt is Npad x Kpad f16,
//   zero padded in both N and K).
// stride: window stride; dil: lhs (input) dilation (2) for ConvTranspose.
// OH*OW and OW must be powers of two (true for every layer in this model);
// lgOHW/lgOW are their log2. gemm_out: store Y row-major M x Cout.
// ---------------------------------------------------------------------------
__global__ __launch_bounds__(TB) void conv_wmma_f16(
    const float* __restrict__ X, const _Float16* __restrict__ Wt,
    const float* __restrict__ bias, float* __restrict__ Y,
    int Nb, int Cin, int H, int W, int Cout,
    int OH, int OW, int lgOHW, int lgOW, int KH, int KW,
    int stride, int pad, int dil, int Kdim, int gemm_out)
{
    __shared__ __align__(16) _Float16 As[2][BM][LDA];
    __shared__ __align__(16) _Float16 Bs[2][BN][LDB];

    const int tid  = threadIdx.x;
    const int lane = tid & 31;
    const int wave = tid >> 5;     // 0..7
    const int wm   = wave & 3;     // M subtile (32 rows)
    const int wn   = wave >> 2;    // N subtile (32 cols)

    const int  OHW  = OH * OW;
    const long Mtot = (long)Nb * OHW;
    const int  m0   = blockIdx.x * BM;
    const int  n0   = blockIdx.y * BN;
    const int  KHW  = KH * KW;
    const int  Kpad = ((Kdim + BK - 1) / BK) * BK;
    const int  nsteps = Kpad / BK;

    // A-tile fill: thread owns one k-column (k decoded ONCE per stage), walks
    // 32 rows. Row decode is shifts/masks (pow2 spatial dims).
    auto loadA = [&](int buf, int k0) {
        const int kk = tid & 63;
        const int mlB = tid >> 6;          // 0..3
        const int k  = k0 + kk;
        const bool kvalid = (k < Kdim);
        int ci = 0, ky = 0, kx = 0;
        if (kvalid) {
            ci = k / KHW;
            int kr = k - ci * KHW;
            ky = kr / KW;
            kx = kr - ky * KW;
        }
        #pragma unroll
        for (int e = 0; e < BM / 4; ++e) {
            int  ml = mlB + e * 4;
            long mg = (long)m0 + ml;
            float v = 0.f;
            if (kvalid && mg < Mtot) {
                int mi   = (int)mg;
                int nimg = mi >> lgOHW;
                int rr   = mi & (OHW - 1);
                int oy   = rr >> lgOW;
                int ox   = rr & (OW - 1);
                int iy   = oy * stride - pad + ky;
                int ix   = ox * stride - pad + kx;
                bool ok  = (iy >= 0) && (ix >= 0);
                if (dil > 1) {             // ConvTranspose: lhs dilation == 2
                    ok = ok && !(iy & 1) && !(ix & 1);
                    iy >>= 1; ix >>= 1;
                }
                if (ok && iy < H && ix < W)
                    v = X[(((size_t)nimg * Cin + ci) * H + iy) * W + ix];
            }
            As[buf][ml][kk] = (_Float16)v;
        }
    };

    // B-tile fill: weights are N/K padded -> branch-free 16B chunks.
    // 512 chunks of 8 f16; async global->LDS when the engine is exposed.
    auto loadB = [&](int buf, int k0) {
        #pragma unroll
        for (int e = 0; e < 2; ++e) {
            int c  = tid + e * TB;
            int nl = c >> 3;
            int kc = (c & 7) * 8;
            const _Float16* g = &Wt[(size_t)(n0 + nl) * Kpad + k0 + kc];
#if defined(USE_ASYNC_LDS)
            __builtin_amdgcn_global_load_async_to_lds_b128(
                (gptr_v4i)(void*)g,
                (lptr_v4i)(void*)&Bs[buf][nl][kc],
                0, 0);
#else
            *(v8h*)&Bs[buf][nl][kc] = *(const v8h*)g;
#endif
        }
    };

    auto fragA = [&](int buf, int mrow, int kk0) -> v16h {
        const int koff = kk0 + (lane >> 4) * 8;
        v8h lo = *(const v8h*)&As[buf][mrow][koff];
        v8h hi = *(const v8h*)&As[buf][mrow][koff + 16];
        v16h f;
        #pragma unroll
        for (int i = 0; i < 8; ++i) { f[i] = lo[i]; f[i + 8] = hi[i]; }
        return f;
    };
    auto fragB = [&](int buf, int nrow, int kk0) -> v16h {
        const int kb = kk0 + (lane >> 4) * 8;
        v8h lo = *(const v8h*)&Bs[buf][nrow][kb];
        v8h hi = *(const v8h*)&Bs[buf][nrow][kb + 16];
        v16h f;
        #pragma unroll
        for (int i = 0; i < 8; ++i) { f[i] = lo[i]; f[i + 8] = hi[i]; }
        return f;
    };

    v8f acc00 = {0.f,0.f,0.f,0.f,0.f,0.f,0.f,0.f};
    v8f acc01 = acc00, acc10 = acc00, acc11 = acc00;

    loadA(0, 0);
    loadB(0, 0);
    wait_async0();
    __syncthreads();

    for (int s = 0; s < nsteps; ++s) {
        const int cur = s & 1;
        if (s + 1 < nsteps) {
            __builtin_prefetch(&Wt[(size_t)(n0 + (tid & 63)) * Kpad + (s + 1) * BK], 0, 1);
            loadA(cur ^ 1, (s + 1) * BK);
            loadB(cur ^ 1, (s + 1) * BK);
        }
        const int lm = lane & 15;
        #pragma unroll
        for (int half = 0; half < 2; ++half) {
            const int kk0 = half * 32;
            v16h a0 = fragA(cur, wm * 32 + lm,      kk0);
            v16h a1 = fragA(cur, wm * 32 + 16 + lm, kk0);
            v16h b0 = fragB(cur, wn * 32 + lm,      kk0);
            v16h b1 = fragB(cur, wn * 32 + 16 + lm, kk0);
            acc00 = __builtin_amdgcn_wmma_f32_16x16x32_f16(false, a0, false, b0,
                                                           (short)0, acc00, false, false);
            acc01 = __builtin_amdgcn_wmma_f32_16x16x32_f16(false, a0, false, b1,
                                                           (short)0, acc01, false, false);
            acc10 = __builtin_amdgcn_wmma_f32_16x16x32_f16(false, a1, false, b0,
                                                           (short)0, acc10, false, false);
            acc11 = __builtin_amdgcn_wmma_f32_16x16x32_f16(false, a1, false, b1,
                                                           (short)0, acc11, false, false);
        }
        wait_async0();
        __syncthreads();
    }

    // --- epilogue: C layout (VGPR r -> M = r + 8*(lane>=16), N = lane&15)
    const int rowHalf = (lane >> 4) * 8;
    const int lcol    = lane & 15;
    #pragma unroll
    for (int mi2 = 0; mi2 < 2; ++mi2) {
        #pragma unroll
        for (int ni = 0; ni < 2; ++ni) {
            const v8f& acc = (mi2 == 0) ? (ni == 0 ? acc00 : acc01)
                                        : (ni == 0 ? acc10 : acc11);
            int col = n0 + wn * 32 + ni * 16 + lcol;
            if (col >= Cout) continue;
            float badd = bias ? bias[col] : 0.f;
            #pragma unroll
            for (int r = 0; r < 8; ++r) {
                long mg = (long)m0 + wm * 32 + mi2 * 16 + rowHalf + r;
                if (mg >= Mtot) continue;
                int mi   = (int)mg;
                int nimg = mi >> lgOHW;
                int rr   = mi & (OHW - 1);
                int oy   = rr >> lgOW;
                int ox   = rr & (OW - 1);
                float o  = acc[r] + badd;
                if (gemm_out) Y[(size_t)mi * Cout + col] = o;
                else          Y[(((size_t)nimg * Cout + col) * OH + oy) * OW + ox] = o;
            }
        }
    }
}

// ---------------------------------------------------------------------------
// Weight prep: f32 (Cout,Cin,KH,KW) -> f16 Wt[Npad][Kpad], zero padded.
// flipT: ConvTranspose — source is (Cin,Cout,KH,KW), 180-degree kernel flip.
// ---------------------------------------------------------------------------
__global__ void prep_weight(const float* __restrict__ w, _Float16* __restrict__ out,
                            int Cout, int Npad, int Cin, int KH, int KW,
                            int Kpad, int flipT)
{
    int i = blockIdx.x * blockDim.x + threadIdx.x;
    int total = Npad * Kpad;
    if (i >= total) return;
    int n = i / Kpad;
    int k = i - n * Kpad;
    int KHW = KH * KW;
    _Float16 v = (_Float16)0.f;
    if (n < Cout && k < Cin * KHW) {
        int ci = k / KHW;
        int r  = k - ci * KHW;
        int ky = r / KW;
        int kx = r - ky * KW;
        float f;
        if (flipT)
            f = w[(((size_t)ci * Cout + n) * KH + (KH - 1 - ky)) * KW + (KW - 1 - kx)];
        else
            f = w[(((size_t)n * Cin + ci) * KH + ky) * KW + kx];
        v = (_Float16)f;
    }
    out[(size_t)n * Kpad + k] = v;
}

__global__ void code_norms(const float* __restrict__ cb, float* __restrict__ cnorm,
                           int K, int D)
{
    int k = blockIdx.x * blockDim.x + threadIdx.x;
    if (k >= K) return;
    float s = 0.f;
    for (int d = 0; d < D; ++d) { float v = cb[(size_t)k * D + d]; s += v * v; }
    cnorm[k] = s;
}

// ---------------------------------------------------------------------------
// Training-mode BatchNorm: per-channel mean / rstd over (N,H,W), then apply.
// ---------------------------------------------------------------------------
__global__ __launch_bounds__(256) void bn_stats(const float* __restrict__ x,
                                                float* __restrict__ mean,
                                                float* __restrict__ rstd,
                                                int Nb, int C, int HW)
{
    int c = blockIdx.x;
    __shared__ float s1[256], s2[256];
    float a = 0.f, b = 0.f;
    long cnt = (long)Nb * HW;
    for (long i = threadIdx.x; i < cnt; i += 256) {
        int n = (int)(i / HW);
        int p = (int)(i - (long)n * HW);
        float v = x[((size_t)n * C + c) * HW + p];
        a += v; b += v * v;
    }
    s1[threadIdx.x] = a; s2[threadIdx.x] = b;
    __syncthreads();
    for (int off = 128; off > 0; off >>= 1) {
        if (threadIdx.x < off) {
            s1[threadIdx.x] += s1[threadIdx.x + off];
            s2[threadIdx.x] += s2[threadIdx.x + off];
        }
        __syncthreads();
    }
    if (threadIdx.x == 0) {
        float m = s1[0] / (float)cnt;
        float v = s2[0] / (float)cnt - m * m;
        mean[c] = m;
        rstd[c] = rsqrtf(v + 1e-5f);
    }
}

__global__ void bn_apply(const float* __restrict__ x, float* __restrict__ y,
                         const float* __restrict__ mean, const float* __restrict__ rstd,
                         const float* __restrict__ g, const float* __restrict__ b,
                         int C, int HW, long total, int relu)
{
    long i = (long)blockIdx.x * blockDim.x + threadIdx.x;
    if (i >= total) return;
    int c = (int)((i / HW) % C);
    float v = (x[i] - mean[c]) * rstd[c] * g[c] + b[c];
    if (relu) v = fmaxf(v, 0.f);
    y[i] = v;
}

__global__ void add_inplace(float* __restrict__ x, const float* __restrict__ h, long total)
{
    long i = (long)blockIdx.x * blockDim.x + threadIdx.x;
    if (i < total) x[i] += h[i];
}

__global__ void zero_f32(float* __restrict__ p, int n)
{
    int i = blockIdx.x * blockDim.x + threadIdx.x;
    if (i < n) p[i] = 0.f;
}

// ---------------------------------------------------------------------------
// VQ: argmin over codes using G = z @ cb^T (d = ||c||^2 - 2 z.c), gather code
// into zq (NCHW), accumulate SSE and code counts.
// ---------------------------------------------------------------------------
__global__ __launch_bounds__(256) void vq_select(
    const float* __restrict__ G, const float* __restrict__ cnorm,
    const float* __restrict__ cb, const float* __restrict__ z,
    float* __restrict__ zq, float* __restrict__ sse, float* __restrict__ counts,
    int Ntok, int Kc, int D, int Hh, int Ww)
{
    int n = blockIdx.x * blockDim.x + threadIdx.x;
    if (n >= Ntok) return;
    const float* g = G + (size_t)n * Kc;
    int best = 0;
    float bv = cnorm[0] - 2.f * g[0];
    for (int k = 1; k < Kc; ++k) {
        float v = cnorm[k] - 2.f * g[k];
        if (v < bv) { bv = v; best = k; }
    }
    int HW   = Hh * Ww;
    int bimg = n / HW;
    int r    = n - bimg * HW;
    int yy   = r / Ww;
    int xx   = r - yy * Ww;
    float local = 0.f;
    for (int d = 0; d < D; ++d) {
        float q = cb[(size_t)best * D + d];
        size_t zi = (((size_t)bimg * D + d) * Hh + yy) * Ww + xx;
        float zz = z[zi];
        zq[zi] = q;                         // straight-through: forward value = q
        float df = q - zz;
        local += df * df;
    }
    atomicAdd(sse, local);
    atomicAdd(&counts[best], 1.f);
}

__global__ __launch_bounds__(256) void vq_finalize(
    const float* __restrict__ sse, const float* __restrict__ counts,
    float* __restrict__ loss_out, float* __restrict__ perp_out,
    int Ntok, int D, int Kc)
{
    __shared__ float sh[256];
    float a = 0.f;
    for (int k = threadIdx.x; k < Kc; k += 256) {
        float avg = counts[k] / (float)Ntok;
        a += avg * logf(avg + 1e-10f);
    }
    sh[threadIdx.x] = a;
    __syncthreads();
    for (int off = 128; off > 0; off >>= 1) {
        if (threadIdx.x < off) sh[threadIdx.x] += sh[threadIdx.x + off];
        __syncthreads();
    }
    if (threadIdx.x == 0) {
        *perp_out = expf(-sh[0]);
        *loss_out = 1.25f * sse[0] / ((float)Ntok * (float)D);
    }
}

// ---------------------------------------------------------------------------
// Host orchestration
// ---------------------------------------------------------------------------
static void run_conv(hipStream_t s, const float* X, const _Float16* Wt,
                     const float* bias, float* Y,
                     int Nb, int Cin, int H, int W, int Cout,
                     int KH, int KW, int stride, int pad, int dil,
                     int OH, int OW, int gemm_out)
{
    long M = (long)Nb * OH * OW;
    int lgOW  = __builtin_ctz(OW);
    int lgOHW = __builtin_ctz(OH * OW);
    dim3 g((unsigned)((M + BM - 1) / BM), (unsigned)((Cout + BN - 1) / BN));
    conv_wmma_f16<<<g, TB, 0, s>>>(X, Wt, bias, Y, Nb, Cin, H, W, Cout,
                                   OH, OW, lgOHW, lgOW, KH, KW,
                                   stride, pad, dil, Cin * KH * KW, gemm_out);
}

enum {
    IN_X = 0,
    E_C1_W, E_C1_B, E_BN1_G, E_BN1_B,
    E_C2_W, E_C2_B, E_BN2_G, E_BN2_B,
    E_C3_W, E_C3_B,
    E_R0_BN1_G, E_R0_BN1_B, E_R0_C1_W, E_R0_BN2_G, E_R0_BN2_B, E_R0_C2_W,
    E_R1_BN1_G, E_R1_BN1_B, E_R1_C1_W, E_R1_BN2_G, E_R1_BN2_B, E_R1_C2_W,
    E_RBN_G, E_RBN_B,
    D_R0_BN1_G, D_R0_BN1_B, D_R0_C1_W, D_R0_BN2_G, D_R0_BN2_B, D_R0_C2_W,
    D_R1_BN1_G, D_R1_BN1_B, D_R1_C1_W, D_R1_BN2_G, D_R1_BN2_B, D_R1_C2_W,
    D_RBN_G, D_RBN_B,
    PRE_W, PRE_B, CODEBOOK,
    D_C1_W, D_C1_B, D_CT1_W, D_CT1_B, D_BN_G, D_BN_B, D_CT2_W, D_CT2_B
};

static inline int rnd64(int v) { return (v + 63) & ~63; }

extern "C" void kernel_launch(void* const* d_in, const int* in_sizes, int n_in,
                              void* d_out, int out_size, void* d_ws, size_t ws_size,
                              hipStream_t stream)
{
    (void)in_sizes; (void)n_in; (void)ws_size;
    auto P = [&](int i) { return (const float*)d_in[i]; };
    const float* x = P(IN_X);

    float* out    = (float*)d_out;
    float* loss_p = out;
    float* recon  = out + 1;
    float* perp_p = out + (out_size - 1);

    // ---- workspace bump allocator ----
    size_t off = 0;
    auto alloc = [&](size_t bytes) -> char* {
        char* p = (char*)d_ws + off;
        off += (bytes + 255) & ~(size_t)255;
        return p;
    };
    float* A1 = (float*)alloc((size_t)16 * 64 * 128 * 128 * 4);   // e1, later ct1 out
    float* A2 = (float*)alloc((size_t)16 * 128 * 64 * 64 * 4);    // e2, later dec stream
    float* A3 = (float*)alloc((size_t)16 * 128 * 64 * 64 * 4);    // e3 / enc res stream
    float* T1 = (float*)alloc((size_t)16 * 128 * 64 * 64 * 4);
    float* T2 = (float*)alloc((size_t)16 * 64 * 64 * 64 * 4);
    float* T3 = (float*)alloc((size_t)16 * 128 * 64 * 64 * 4);
    float* Z  = (float*)alloc((size_t)16 * 64 * 64 * 64 * 4);
    float* G  = (float*)alloc((size_t)65536 * 512 * 4);
    float* ZQ = (float*)alloc((size_t)16 * 64 * 64 * 64 * 4);
    float* meanb  = (float*)alloc(512 * 4);
    float* rstdb  = (float*)alloc(512 * 4);
    float* cnorm  = (float*)alloc(512 * 4);
    float* sse    = (float*)alloc(4);
    float* counts = (float*)alloc(512 * 4);

    auto allocW = [&](int Cout, int Cin, int KH, int KW) -> _Float16* {
        int Kp = rnd64(Cin * KH * KW);
        int Np = rnd64(Cout);
        return (_Float16*)alloc((size_t)Np * Kp * 2);
    };
    _Float16* w_ec1 = allocW(64, 3, 4, 4);
    _Float16* w_ec2 = allocW(128, 64, 4, 4);
    _Float16* w_ec3 = allocW(128, 128, 3, 3);
    _Float16* w_erc1[2] = { allocW(64, 128, 3, 3), allocW(64, 128, 3, 3) };
    _Float16* w_erc2[2] = { allocW(128, 64, 1, 1), allocW(128, 64, 1, 1) };
    _Float16* w_drc1[2] = { allocW(64, 128, 3, 3), allocW(64, 128, 3, 3) };
    _Float16* w_drc2[2] = { allocW(128, 64, 1, 1), allocW(128, 64, 1, 1) };
    _Float16* w_pre = allocW(64, 128, 1, 1);
    _Float16* w_cb  = allocW(512, 64, 1, 1);
    _Float16* w_dc1 = allocW(128, 64, 3, 3);
    _Float16* w_ct1 = allocW(64, 128, 4, 4);
    _Float16* w_ct2 = allocW(3, 64, 4, 4);

    auto prep = [&](int widx, _Float16* dst, int Cout, int Cin, int KH, int KW, int flip) {
        int Kp  = rnd64(Cin * KH * KW);
        int Np  = rnd64(Cout);
        int tot = Np * Kp;
        prep_weight<<<(tot + 255) / 256, 256, 0, stream>>>(P(widx), dst, Cout, Np,
                                                           Cin, KH, KW, Kp, flip);
    };
    prep(E_C1_W, w_ec1, 64, 3, 4, 4, 0);
    prep(E_C2_W, w_ec2, 128, 64, 4, 4, 0);
    prep(E_C3_W, w_ec3, 128, 128, 3, 3, 0);
    prep(E_R0_C1_W, w_erc1[0], 64, 128, 3, 3, 0);
    prep(E_R0_C2_W, w_erc2[0], 128, 64, 1, 1, 0);
    prep(E_R1_C1_W, w_erc1[1], 64, 128, 3, 3, 0);
    prep(E_R1_C2_W, w_erc2[1], 128, 64, 1, 1, 0);
    prep(D_R0_C1_W, w_drc1[0], 64, 128, 3, 3, 0);
    prep(D_R0_C2_W, w_drc2[0], 128, 64, 1, 1, 0);
    prep(D_R1_C1_W, w_drc1[1], 64, 128, 3, 3, 0);
    prep(D_R1_C2_W, w_drc2[1], 128, 64, 1, 1, 0);
    prep(PRE_W,  w_pre, 64, 128, 1, 1, 0);
    prep(CODEBOOK, w_cb, 512, 64, 1, 1, 0);       // (K,D) is already N x K layout
    prep(D_C1_W, w_dc1, 128, 64, 3, 3, 0);
    prep(D_CT1_W, w_ct1, 64, 128, 4, 4, 1);       // (in=128,out=64,4,4), flipped
    prep(D_CT2_W, w_ct2, 3, 64, 4, 4, 1);         // (in=64,out=3,4,4), flipped
    code_norms<<<2, 256, 0, stream>>>(P(CODEBOOK), cnorm, 512, 64);
    zero_f32<<<3, 256, 0, stream>>>(counts, 512);
    zero_f32<<<1, 32, 0, stream>>>(sse, 1);

    auto bn_relu = [&](float* Xb, int gi, int bi, int C, int HW, int relu) {
        long tot = (long)16 * C * HW;
        bn_stats<<<C, 256, 0, stream>>>(Xb, meanb, rstdb, 16, C, HW);
        bn_apply<<<(unsigned)((tot + 255) / 256), 256, 0, stream>>>(
            Xb, Xb, meanb, rstdb, P(gi), P(bi), C, HW, tot, relu);
    };

    auto res_stack = [&](float* Xb, int base, _Float16* const* wc1,
                         _Float16* const* wc2, int rbn) {
        const long tot128 = (long)16 * 128 * 4096;
        const long tot64  = (long)16 * 64 * 4096;
        for (int i = 0; i < 2; ++i) {
            int b0 = base + i * 6;
            bn_stats<<<128, 256, 0, stream>>>(Xb, meanb, rstdb, 16, 128, 4096);
            bn_apply<<<(unsigned)((tot128 + 255) / 256), 256, 0, stream>>>(
                Xb, T1, meanb, rstdb, P(b0 + 0), P(b0 + 1), 128, 4096, tot128, 1);
            run_conv(stream, T1, wc1[i], nullptr, T2, 16, 128, 64, 64, 64,
                     3, 3, 1, 1, 1, 64, 64, 0);
            bn_stats<<<64, 256, 0, stream>>>(T2, meanb, rstdb, 16, 64, 4096);
            bn_apply<<<(unsigned)((tot64 + 255) / 256), 256, 0, stream>>>(
                T2, T2, meanb, rstdb, P(b0 + 3), P(b0 + 4), 64, 4096, tot64, 1);
            run_conv(stream, T2, wc2[i], nullptr, T3, 16, 64, 64, 64, 128,
                     1, 1, 1, 0, 1, 64, 64, 0);
            add_inplace<<<(unsigned)((tot128 + 255) / 256), 256, 0, stream>>>(Xb, T3, tot128);
        }
        bn_stats<<<128, 256, 0, stream>>>(Xb, meanb, rstdb, 16, 128, 4096);
        bn_apply<<<(unsigned)((tot128 + 255) / 256), 256, 0, stream>>>(
            Xb, Xb, meanb, rstdb, P(rbn), P(rbn + 1), 128, 4096, tot128, 1);
    };

    // ---------------- encoder ----------------
    run_conv(stream, x,  w_ec1, P(E_C1_B), A1, 16, 3, 256, 256, 64, 4, 4, 2, 1, 1, 128, 128, 0);
    bn_relu(A1, E_BN1_G, E_BN1_B, 64, 128 * 128, 1);
    run_conv(stream, A1, w_ec2, P(E_C2_B), A2, 16, 64, 128, 128, 128, 4, 4, 2, 1, 1, 64, 64, 0);
    bn_relu(A2, E_BN2_G, E_BN2_B, 128, 64 * 64, 1);
    run_conv(stream, A2, w_ec3, P(E_C3_B), A3, 16, 128, 64, 64, 128, 3, 3, 1, 1, 1, 64, 64, 0);
    res_stack(A3, E_R0_BN1_G, w_erc1, w_erc2, E_RBN_G);
    run_conv(stream, A3, w_pre, P(PRE_B), Z, 16, 128, 64, 64, 64, 1, 1, 1, 0, 1, 64, 64, 0);

    // ---------------- vector quantizer ----------------
    run_conv(stream, Z, w_cb, nullptr, G, 16, 64, 64, 64, 512, 1, 1, 1, 0, 1, 64, 64, 1);
    vq_select<<<(65536 + 255) / 256, 256, 0, stream>>>(G, cnorm, P(CODEBOOK), Z,
                                                       ZQ, sse, counts,
                                                       65536, 512, 64, 64, 64);
    vq_finalize<<<1, 256, 0, stream>>>(sse, counts, loss_p, perp_p, 65536, 64, 512);

    // ---------------- decoder ----------------
    run_conv(stream, ZQ, w_dc1, P(D_C1_B), A2, 16, 64, 64, 64, 128, 3, 3, 1, 1, 1, 64, 64, 0);
    res_stack(A2, D_R0_BN1_G, w_drc1, w_drc2, D_RBN_G);
    run_conv(stream, A2, w_ct1, P(D_CT1_B), A1, 16, 128, 64, 64, 64, 4, 4, 1, 2, 2, 128, 128, 0);
    bn_relu(A1, D_BN_G, D_BN_B, 64, 128 * 128, 1);
    run_conv(stream, A1, w_ct2, P(D_CT2_B), recon, 16, 64, 128, 128, 3, 4, 4, 1, 2, 2, 256, 256, 0);
}
